// PaiNNEncoder_66288525246595
// MI455X (gfx1250) — matrix operand order repaired and checked by reading, not compiled, not run on hardware
//
#include <hip/hip_runtime.h>
#include <hip/hip_bf16.h>
#include <math.h>

typedef __attribute__((ext_vector_type(16))) __bf16 v16bf;
typedef __attribute__((ext_vector_type(8)))  __bf16 v8bf;
typedef __attribute__((ext_vector_type(2)))  __bf16 v2bf;
typedef __attribute__((ext_vector_type(8)))  float  v8f;

#define HD   128
#define NN   20000
#define NE   160000
#define EDIM 19
#define NBLK 6
#define RSC  0.1f
#define CLV  50.0f

// ---------------------------------------------------------------------------
// WMMA helpers (bf16 in, f32 accumulate), layouts per CDNA5 ISA 7.12.2
// ---------------------------------------------------------------------------
__device__ __forceinline__ v8f wmma_bf16(v16bf a, v16bf b, v8f c) {
    return __builtin_amdgcn_wmma_f32_16x16x32_bf16(false, a, false, b, (short)0, c,
                                                   false, false);
}

// A-matrix 16x32 bf16 fragment from LDS (row-major, stride lda elements).
// Lane's 16 values = two contiguous 8-element (16B) chunks:
//   [kbase + (l>>4)*8, +8) and [kbase + 16 + (l>>4)*8, +8), row M = lane&15.
// All callers keep rows 16B-aligned so these are two ds_load_b128.
__device__ __forceinline__ v16bf frag_a_lds(const __bf16* base, int lda, int kbase,
                                            int lane) {
    const __bf16* rowp = base + (lane & 15) * lda + kbase + ((lane >> 4) * 8);
    v8bf lo = *(const v8bf*)rowp;
    v8bf hi = *(const v8bf*)(rowp + 16);
    return __builtin_shufflevector(lo, hi, 0, 1, 2, 3, 4, 5, 6, 7, 8, 9, 10, 11, 12,
                                   13, 14, 15);
}

// B-matrix fragment from pre-swizzled packed bf16 weights: per (nt,kt) fragment,
// lane's 16 values are contiguous (32B, 32B-aligned) -> one v16bf load.
__device__ __forceinline__ v16bf frag_b_pk(const __bf16* __restrict__ P, int nKt,
                                           int nt, int kt, int lane) {
    const __bf16* p = P + ((((size_t)nt * nKt + kt) * 32 + lane) << 4);
    return *(const v16bf*)p;
}

// SiLU with hardware reciprocal (v_rcp_f32) instead of IEEE divide.
__device__ __forceinline__ float silu_f(float x) {
    return x * __builtin_amdgcn_rcpf(1.0f + __expf(-x));
}
__device__ __forceinline__ float clamp_f(float x) { return fminf(fmaxf(x, -CLV), CLV); }

// ---------------------------------------------------------------------------
// Weight repack: W[Nw,Kw] f32 row-major -> packed bf16 fragments.
// dst[((nt*nKt+kt)*32 + l)*16 + e] = bf16(W[nt*16+(l&15), kt*32+(e>>3)*16+(l>>4)*8+(e&7)])
// (zero beyond Kact, K padded to Kpad multiple of 32)
// ---------------------------------------------------------------------------
__global__ void repack_w_kernel(const float* __restrict__ W, __bf16* __restrict__ dst,
                                int Nw, int Kw, int Kpad, int Kact) {
    int idx = blockIdx.x * blockDim.x + threadIdx.x;
    if (idx >= Nw * Kpad) return;
    int nKt = Kpad >> 5;
    int f = idx >> 9, w = idx & 511;
    int l = w >> 4, e = w & 15;
    int nt = f / nKt, kt = f - nt * nKt;
    int n = nt * 16 + (l & 15);
    int k = kt * 32 + ((e >> 3) * 16) + ((l >> 4) * 8) + (e & 7);
    float val = (k < Kact) ? W[(size_t)n * Kw + k] : 0.0f;
    dst[idx] = (__bf16)val;
}

// ---------------------------------------------------------------------------
// Init kernels
// ---------------------------------------------------------------------------
__global__ void init_s_kernel(const int* __restrict__ z, const float* __restrict__ emb,
                              float* __restrict__ s, __bf16* __restrict__ s_bf) {
    int i = blockIdx.x * blockDim.x + threadIdx.x;
    if (i < NN * HD) {
        int n = i >> 7, h = i & 127;
        float val = emb[(size_t)z[n] * HD + h];
        s[i] = val;
        s_bf[i] = (__bf16)val;
    }
}

__global__ void init_v_kernel(const float* __restrict__ vf,
                              const float* __restrict__ vproj_w, float* __restrict__ v) {
    int i = blockIdx.x * blockDim.x + threadIdx.x;
    if (i < NN * HD) {
        int n = i >> 7, h = i & 127;
        float val = vf[n * 3 + 0] * vproj_w[h * 3 + 0] +
                    vf[n * 3 + 1] * vproj_w[h * 3 + 1] +
                    vf[n * 3 + 2] * vproj_w[h * 3 + 2];
        size_t o = (size_t)n * (HD * 3) + h * 3;
        v[o + 0] = val; v[o + 1] = val; v[o + 2] = val;
    }
}

__global__ void init_evec_kernel(const int* __restrict__ ei, const float* __restrict__ pos,
                                 float* __restrict__ evec) {
    int e = blockIdx.x * blockDim.x + threadIdx.x;
    if (e < NE) {
        int r = ei[e];       // row = source j
        int c = ei[NE + e];  // col = target i
#pragma unroll
        for (int d = 0; d < 3; ++d)
            evec[(size_t)e * 3 + d] = pos[(size_t)c * 3 + d] - pos[(size_t)r * 3 + d];
    }
}

__global__ void zero2_kernel(float* __restrict__ a, long na, float* __restrict__ b,
                             long nb) {
    long i = (long)blockIdx.x * blockDim.x + threadIdx.x;
    long stride = (long)gridDim.x * blockDim.x;
    for (long j = i; j < na; j += stride) a[j] = 0.0f;
    for (long j = i; j < nb; j += stride) b[j] = 0.0f;
}

__global__ void copy_v_kernel(const float* __restrict__ v, float* __restrict__ out) {
    long i = (long)blockIdx.x * blockDim.x + threadIdx.x;
    if (i < (long)NN * HD * 3) out[i] = v[i];
}

// ---------------------------------------------------------------------------
// Edge message kernel: 16 edges per workgroup (128 threads = 4 waves)
// ---------------------------------------------------------------------------
struct EdgeSmem {
    __bf16 As[16][264];   // cat(s_i, s_j), K=256 (+8 pad), rows 16B aligned
    __bf16 Ae[16][40];    // edge_attr padded 19 -> 32
    __bf16 h1[16][136];   // silu(s1(...))
    __bf16 he[16][136];   // silu(e1(...))
    float  phi[16][392];  // phi * pe, 3H = 384 (+8 pad)
    float  evec[16][3];
    int    ci[16], ri[16];
};

__global__ __launch_bounds__(128) void edge_msg_kernel(
    const __bf16* __restrict__ s_bf, const float* __restrict__ v,
    const float* __restrict__ evec_g, const int* __restrict__ rowi,
    const int* __restrict__ coli, const float* __restrict__ eattr,
    const __bf16* __restrict__ s1p, const float* __restrict__ s1_b,
    const __bf16* __restrict__ s2p, const float* __restrict__ s2_b,
    const __bf16* __restrict__ e1p, const float* __restrict__ e1_b,
    const __bf16* __restrict__ e2p, const float* __restrict__ e2_b,
    float* __restrict__ ds_acc, float* __restrict__ dv_acc) {
    __shared__ EdgeSmem sm;
    const int tid = threadIdx.x, lane = tid & 31, wave = tid >> 5;
    const int e0 = blockIdx.x * 16;

    if (tid < 16) {
        sm.ci[tid] = coli[e0 + tid];
        sm.ri[tid] = rowi[e0 + tid];
    }
    if (tid >= 64 && tid < 112) {
        int t = tid - 64;
        sm.evec[t / 3][t % 3] = evec_g[(size_t)(e0 + t / 3) * 3 + (t % 3)];
    }
    __syncthreads();

    // gather A = cat(s_bf[col], s_bf[row]) as 16B chunks (no conversion)
    for (int idx = tid; idx < 16 * 32; idx += 128) {
        int el = idx >> 5, ch = idx & 31;
        int node = (ch < 16) ? sm.ci[el] : sm.ri[el];
        const v8bf* src = (const v8bf*)(s_bf + (size_t)node * HD) + (ch & 15);
        *((v8bf*)(&sm.As[el][0]) + ch) = *src;
    }
    // gather edge_attr (zero-pad to 32), packed pair stores
    for (int idx = tid; idx < 16 * 16; idx += 128) {
        int el = idx >> 4, kp = (idx & 15) * 2;
        float x0 = (kp < EDIM) ? eattr[(size_t)(e0 + el) * EDIM + kp] : 0.0f;
        float x1 = (kp + 1 < EDIM) ? eattr[(size_t)(e0 + el) * EDIM + kp + 1] : 0.0f;
        v2bf pk; pk[0] = (__bf16)x0; pk[1] = (__bf16)x1;
        *(v2bf*)&sm.Ae[el][kp] = pk;
    }
    __syncthreads();

    // h1 = silu(As @ s1_w^T + b)  [16 x 128], K=256
    {
        v16bf afr[8];
#pragma unroll
        for (int kt = 0; kt < 8; ++kt)
            afr[kt] = frag_a_lds(&sm.As[0][0], 264, kt * 32, lane);
        for (int t = wave; t < 8; t += 4) {
            v8f acc = {};
#pragma unroll
            for (int kt = 0; kt < 8; ++kt)
                acc = wmma_bf16(afr[kt], frag_b_pk(s1p, 8, t, kt, lane), acc);
            int n = t * 16 + (lane & 15);
            float bias = s1_b[n];
#pragma unroll
            for (int r = 0; r < 8; ++r) {
                int m = r + ((lane >> 4) * 8);
                sm.h1[m][n] = (__bf16)silu_f(acc[r] + bias);
            }
        }
    }
    // he = silu(Ae @ e1_w^T + b)  [16 x 128], K padded 19->32
    {
        v16bf ae = frag_a_lds(&sm.Ae[0][0], 40, 0, lane);
        for (int t = wave; t < 8; t += 4) {
            v8f acc = {};
            acc = wmma_bf16(ae, frag_b_pk(e1p, 1, t, 0, lane), acc);
            int n = t * 16 + (lane & 15);
            float bias = e1_b[n];
#pragma unroll
            for (int r = 0; r < 8; ++r) {
                int m = r + ((lane >> 4) * 8);
                sm.he[m][n] = (__bf16)silu_f(acc[r] + bias);
            }
        }
    }
    __syncthreads();

    // phi = (h1 @ s2^T + b) * (he @ e2^T + b)   [16 x 384]
    {
        v16bf ah[4], ag[4];
#pragma unroll
        for (int kt = 0; kt < 4; ++kt) {
            ah[kt] = frag_a_lds(&sm.h1[0][0], 136, kt * 32, lane);
            ag[kt] = frag_a_lds(&sm.he[0][0], 136, kt * 32, lane);
        }
        for (int t = wave; t < 24; t += 4) {
            v8f ap = {}, pp = {};
#pragma unroll
            for (int kt = 0; kt < 4; ++kt) {
                ap = wmma_bf16(ah[kt], frag_b_pk(s2p, 4, t, kt, lane), ap);
                pp = wmma_bf16(ag[kt], frag_b_pk(e2p, 4, t, kt, lane), pp);
            }
            int n = t * 16 + (lane & 15);
            float bs = s2_b[n], be = e2_b[n];
#pragma unroll
            for (int r = 0; r < 8; ++r) {
                int m = r + ((lane >> 4) * 8);
                sm.phi[m][n] = (ap[r] + bs) * (pp[r] + be);
            }
        }
    }
    __syncthreads();

    // scatter: ds[col] += phi1 ; dv[col] += phi2*v[row] + phi3*edge_vec
    for (int idx = tid; idx < 16 * HD; idx += 128) {
        int el = idx >> 7, h = idx & 127;
        int ct = sm.ci[el], rt = sm.ri[el];
        float p1 = sm.phi[el][h];
        float p2 = sm.phi[el][HD + h];
        float p3 = sm.phi[el][2 * HD + h];
        atomicAdd(&ds_acc[(size_t)ct * HD + h], p1);
#pragma unroll
        for (int c = 0; c < 3; ++c) {
            float vm = v[(size_t)rt * (HD * 3) + h * 3 + c];
            atomicAdd(&dv_acc[(size_t)ct * (HD * 3) + h * 3 + c],
                      p2 * vm + p3 * sm.evec[el][c]);
        }
    }
}

// ---------------------------------------------------------------------------
// Node update kernel: 16 nodes per workgroup (128 threads = 4 waves)
// ---------------------------------------------------------------------------
struct NodeSmem {
    __bf16 Ads[16][136];     // ds accumulator (bf16)
    __bf16 Adv[3][16][136];  // dv accumulator per channel
    __bf16 Av[3][16][136];   // v_new per channel
    __bf16 Acat[16][264];    // [s_new | vn] for u1 GEMM
    __bf16 hu[16][136];      // silu(u1(...))
};

__global__ __launch_bounds__(128) void node_update_kernel(
    float* __restrict__ s, __bf16* __restrict__ s_bf, float* __restrict__ v,
    const float* __restrict__ ds_acc, const float* __restrict__ dv_acc,
    const __bf16* __restrict__ fsp, const float* __restrict__ fs_b,
    const __bf16* __restrict__ fvp,
    const __bf16* __restrict__ Up, const __bf16* __restrict__ Vp,
    const __bf16* __restrict__ u1p, const float* __restrict__ u1_b,
    const __bf16* __restrict__ u2p, const float* __restrict__ u2_b) {
    __shared__ NodeSmem sm;
    const int tid = threadIdx.x, lane = tid & 31, wave = tid >> 5;
    const int n0 = blockIdx.x * 16;

    for (int idx = tid; idx < 16 * 64; idx += 128) {
        int m = idx >> 6, kp = (idx & 63) * 2;
        const float* dsp = &ds_acc[(size_t)(n0 + m) * HD + kp];
        v2bf pk; pk[0] = (__bf16)dsp[0]; pk[1] = (__bf16)dsp[1];
        *(v2bf*)&sm.Ads[m][kp] = pk;
        const float* dvp = &dv_acc[(size_t)(n0 + m) * (HD * 3) + kp * 3];
#pragma unroll
        for (int c = 0; c < 3; ++c) {
            v2bf pv; pv[0] = (__bf16)dvp[c]; pv[1] = (__bf16)dvp[3 + c];
            *(v2bf*)&sm.Adv[c][m][kp] = pv;
        }
    }
    __syncthreads();

    float sn_reg[2][8];      // s_new
    float uv_reg[2][3][8];   // Uv
    float vn_reg[2][8];      // ||Vv||

    // s_new = s + RS * fs(ds)  (kept in regs + Acat bf16)
    {
        v16bf af[4];
#pragma unroll
        for (int kt = 0; kt < 4; ++kt)
            af[kt] = frag_a_lds(&sm.Ads[0][0], 136, kt * 32, lane);
#pragma unroll
        for (int j = 0; j < 2; ++j) {
            int t = wave + 4 * j;
            v8f acc = {};
#pragma unroll
            for (int kt = 0; kt < 4; ++kt)
                acc = wmma_bf16(af[kt], frag_b_pk(fsp, 4, t, kt, lane), acc);
            int h = t * 16 + (lane & 15);
            float bias = fs_b[h];
#pragma unroll
            for (int r = 0; r < 8; ++r) {
                int m = r + ((lane >> 4) * 8);
                float sn = s[(size_t)(n0 + m) * HD + h] + RSC * (acc[r] + bias);
                sn_reg[j][r] = sn;
                sm.Acat[m][h] = (__bf16)sn;
            }
        }
    }
    // v_new = v + RS * fv(dv) per channel: write to global v + Av bf16
#pragma unroll
    for (int c = 0; c < 3; ++c) {
        v16bf af[4];
#pragma unroll
        for (int kt = 0; kt < 4; ++kt)
            af[kt] = frag_a_lds(&sm.Adv[c][0][0], 136, kt * 32, lane);
#pragma unroll
        for (int j = 0; j < 2; ++j) {
            int t = wave + 4 * j;
            v8f acc = {};
#pragma unroll
            for (int kt = 0; kt < 4; ++kt)
                acc = wmma_bf16(af[kt], frag_b_pk(fvp, 4, t, kt, lane), acc);
            int h = t * 16 + (lane & 15);
#pragma unroll
            for (int r = 0; r < 8; ++r) {
                int m = r + ((lane >> 4) * 8);
                size_t off = (size_t)(n0 + m) * (HD * 3) + h * 3 + c;
                float vn_ = v[off] + RSC * acc[r];
                v[off] = vn_;               // re-read in final phase (same thread)
                sm.Av[c][m][h] = (__bf16)vn_;
            }
        }
    }
    __syncthreads();

    // Vv/Uv GEMMs; vn = sqrt(sum_c Vv^2)
    {
        float vn2[2][8];
#pragma unroll
        for (int j = 0; j < 2; ++j)
#pragma unroll
            for (int r = 0; r < 8; ++r) vn2[j][r] = 0.0f;
#pragma unroll
        for (int c = 0; c < 3; ++c) {
            v16bf af[4];
#pragma unroll
            for (int kt = 0; kt < 4; ++kt)
                af[kt] = frag_a_lds(&sm.Av[c][0][0], 136, kt * 32, lane);
#pragma unroll
            for (int j = 0; j < 2; ++j) {
                int t = wave + 4 * j;
                v8f accV = {}, accU = {};
#pragma unroll
                for (int kt = 0; kt < 4; ++kt) {
                    accV = wmma_bf16(af[kt], frag_b_pk(Vp, 4, t, kt, lane), accV);
                    accU = wmma_bf16(af[kt], frag_b_pk(Up, 4, t, kt, lane), accU);
                }
#pragma unroll
                for (int r = 0; r < 8; ++r) {
                    vn2[j][r] += accV[r] * accV[r];
                    uv_reg[j][c][r] = accU[r];
                }
            }
        }
#pragma unroll
        for (int j = 0; j < 2; ++j) {
            int t = wave + 4 * j;
            int h = t * 16 + (lane & 15);
#pragma unroll
            for (int r = 0; r < 8; ++r) {
                int m = r + ((lane >> 4) * 8);
                float vnv = sqrtf(vn2[j][r]);
                vn_reg[j][r] = vnv;
                sm.Acat[m][HD + h] = (__bf16)vnv;
            }
        }
    }
    __syncthreads();

    // hu = silu(u1(cat(s_new, vn)))
    {
        v16bf af[8];
#pragma unroll
        for (int kt = 0; kt < 8; ++kt)
            af[kt] = frag_a_lds(&sm.Acat[0][0], 264, kt * 32, lane);
        for (int t = wave; t < 8; t += 4) {
            v8f acc = {};
#pragma unroll
            for (int kt = 0; kt < 8; ++kt)
                acc = wmma_bf16(af[kt], frag_b_pk(u1p, 8, t, kt, lane), acc);
            int h = t * 16 + (lane & 15);
            float bias = u1_b[h];
#pragma unroll
            for (int r = 0; r < 8; ++r) {
                int m = r + ((lane >> 4) * 8);
                sm.hu[m][h] = (__bf16)silu_f(acc[r] + bias);
            }
        }
    }
    __syncthreads();

    // a = u2(hu); final s/v updates with clamp
    {
        v16bf af[4];
#pragma unroll
        for (int kt = 0; kt < 4; ++kt)
            af[kt] = frag_a_lds(&sm.hu[0][0], 136, kt * 32, lane);
#pragma unroll
        for (int j = 0; j < 2; ++j) {
            int t = wave + 4 * j;
            v8f a1 = {}, a2 = {}, a3 = {};
#pragma unroll
            for (int kt = 0; kt < 4; ++kt) {
                a1 = wmma_bf16(af[kt], frag_b_pk(u2p, 4, t, kt, lane), a1);
                a2 = wmma_bf16(af[kt], frag_b_pk(u2p, 4, t + 8, kt, lane), a2);
                a3 = wmma_bf16(af[kt], frag_b_pk(u2p, 4, t + 16, kt, lane), a3);
            }
            int h = t * 16 + (lane & 15);
            float b1 = u2_b[h], b2 = u2_b[HD + h], b3 = u2_b[2 * HD + h];
#pragma unroll
            for (int r = 0; r < 8; ++r) {
                int m = r + ((lane >> 4) * 8);
                float A1 = a1[r] + b1;
                float A2 = a2[r] + b2;
                float A3 = a3[r] + b3;
                size_t si = (size_t)(n0 + m) * HD + h;
                float sv = clamp_f(sn_reg[j][r] + RSC * (A1 + A2 * vn_reg[j][r]));
                s[si] = sv;
                s_bf[si] = (__bf16)sv;
#pragma unroll
                for (int c = 0; c < 3; ++c) {
                    size_t off = (size_t)(n0 + m) * (HD * 3) + h * 3 + c;
                    v[off] = clamp_f(v[off] + RSC * A3 * uv_reg[j][c][r]);
                }
            }
        }
    }
}

// ---------------------------------------------------------------------------
// Output MLP: s_out = out2(silu(out1(s)))
// ---------------------------------------------------------------------------
struct OutSmem {
    __bf16 As[16][136];
    __bf16 h[16][136];
};

__global__ __launch_bounds__(128) void out_mlp_kernel(
    const __bf16* __restrict__ s_bf, const __bf16* __restrict__ w1p,
    const float* __restrict__ b1, const __bf16* __restrict__ w2p,
    const float* __restrict__ b2, float* __restrict__ outp) {
    __shared__ OutSmem sm;
    const int tid = threadIdx.x, lane = tid & 31, wave = tid >> 5;
    const int n0 = blockIdx.x * 16;

    for (int idx = tid; idx < 16 * 16; idx += 128) {  // 16 rows x 16 chunks of 8 bf16
        int m = idx >> 4, ch = idx & 15;
        *((v8bf*)(&sm.As[m][0]) + ch) =
            *((const v8bf*)(s_bf + (size_t)(n0 + m) * HD) + ch);
    }
    __syncthreads();
    {
        v16bf af[4];
#pragma unroll
        for (int kt = 0; kt < 4; ++kt)
            af[kt] = frag_a_lds(&sm.As[0][0], 136, kt * 32, lane);
        for (int t = wave; t < 8; t += 4) {
            v8f acc = {};
#pragma unroll
            for (int kt = 0; kt < 4; ++kt)
                acc = wmma_bf16(af[kt], frag_b_pk(w1p, 4, t, kt, lane), acc);
            int h = t * 16 + (lane & 15);
            float bias = b1[h];
#pragma unroll
            for (int r = 0; r < 8; ++r) {
                int m = r + ((lane >> 4) * 8);
                sm.h[m][h] = (__bf16)silu_f(acc[r] + bias);
            }
        }
    }
    __syncthreads();
    {
        v16bf af[4];
#pragma unroll
        for (int kt = 0; kt < 4; ++kt)
            af[kt] = frag_a_lds(&sm.h[0][0], 136, kt * 32, lane);
        for (int t = wave; t < 8; t += 4) {
            v8f acc = {};
#pragma unroll
            for (int kt = 0; kt < 4; ++kt)
                acc = wmma_bf16(af[kt], frag_b_pk(w2p, 4, t, kt, lane), acc);
            int h = t * 16 + (lane & 15);
            float bias = b2[h];
#pragma unroll
            for (int r = 0; r < 8; ++r) {
                int m = r + ((lane >> 4) * 8);
                outp[(size_t)(n0 + m) * HD + h] = acc[r] + bias;
            }
        }
    }
}

// ---------------------------------------------------------------------------
// Host launch
// ---------------------------------------------------------------------------
extern "C" void kernel_launch(void* const* d_in, const int* in_sizes, int n_in,
                              void* d_out, int out_size, void* d_ws, size_t ws_size,
                              hipStream_t stream) {
    (void)in_sizes; (void)n_in; (void)out_size; (void)ws_size;
    // jax pytree flatten order (dict keys sorted):
    const int*   z       = (const int*)d_in[0];
    const float* vf      = (const float*)d_in[1];
    const int*   ei      = (const int*)d_in[2];
    const float* eattr   = (const float*)d_in[3];
    const float* pos     = (const float*)d_in[4];
    const float* Uw_a    = (const float*)d_in[5];
    const float* Vw_a    = (const float*)d_in[6];
    const float* e1b_a   = (const float*)d_in[7];
    const float* e1w_a   = (const float*)d_in[8];
    const float* e2b_a   = (const float*)d_in[9];
    const float* e2w_a   = (const float*)d_in[10];
    const float* fsb_a   = (const float*)d_in[11];
    const float* fsw_a   = (const float*)d_in[12];
    const float* fvw_a   = (const float*)d_in[13];
    const float* s1b_a   = (const float*)d_in[14];
    const float* s1w_a   = (const float*)d_in[15];
    const float* s2b_a   = (const float*)d_in[16];
    const float* s2w_a   = (const float*)d_in[17];
    const float* u1b_a   = (const float*)d_in[18];
    const float* u1w_a   = (const float*)d_in[19];
    const float* u2b_a   = (const float*)d_in[20];
    const float* u2w_a   = (const float*)d_in[21];
    const float* emb     = (const float*)d_in[22];
    const float* out1_b  = (const float*)d_in[23];
    const float* out1_w  = (const float*)d_in[24];
    const float* out2_b  = (const float*)d_in[25];
    const float* out2_w  = (const float*)d_in[26];
    const float* vproj_w = (const float*)d_in[27];

    // f32 state regions
    float* ws   = (float*)d_ws;
    float* s    = ws;                       // N*H
    float* v    = s + (size_t)NN * HD;      // N*H*3
    float* evec = v + (size_t)NN * HD * 3;  // E*3
    float* dsA  = evec + (size_t)NE * 3;    // N*H
    float* dvA  = dsA + (size_t)NN * HD;    // N*H*3
    // bf16 regions (all offsets keep 32B alignment)
    __bf16* bfp  = (__bf16*)(dvA + (size_t)NN * HD * 3);
    __bf16* s_bf = bfp;  bfp += (size_t)NN * HD;
    // packed weights per block
    const int S1P = 128 * 256, S2P = 384 * 128, E1P = 128 * 32, E2P = 384 * 128;
    const int FSP = 128 * 128, FVP = 128 * 128, UP = 128 * 128, VP = 128 * 128;
    const int U1P = 128 * 256, U2P = 384 * 128;
    const int TPB = S1P + S2P + E1P + E2P + FSP + FVP + UP + VP + U1P + U2P;
    __bf16* blkp = bfp;  bfp += (size_t)TPB * NBLK;
    __bf16* o1p = bfp;   bfp += 128 * 128;
    __bf16* o2p = bfp;

    auto rp = [&](const float* W, __bf16* dst, int Nw, int Kw, int Kpad, int Kact) {
        int total = Nw * Kpad;
        repack_w_kernel<<<(total + 255) / 256, 256, 0, stream>>>(W, dst, Nw, Kw, Kpad,
                                                                 Kact);
    };

    for (int blk = 0; blk < NBLK; ++blk) {
        __bf16* p = blkp + (size_t)blk * TPB;
        rp(s1w_a + (size_t)blk * 128 * 256, p, 128, 256, 256, 256);            p += S1P;
        rp(s2w_a + (size_t)blk * 384 * 128, p, 384, 128, 128, 128);            p += S2P;
        rp(e1w_a + (size_t)blk * 128 * EDIM, p, 128, EDIM, 32, EDIM);          p += E1P;
        rp(e2w_a + (size_t)blk * 384 * 128, p, 384, 128, 128, 128);            p += E2P;
        rp(fsw_a + (size_t)blk * 128 * 128, p, 128, 128, 128, 128);            p += FSP;
        rp(fvw_a + (size_t)blk * 128 * 128, p, 128, 128, 128, 128);            p += FVP;
        rp(Uw_a + (size_t)blk * 128 * 128, p, 128, 128, 128, 128);             p += UP;
        rp(Vw_a + (size_t)blk * 128 * 128, p, 128, 128, 128, 128);             p += VP;
        rp(u1w_a + (size_t)blk * 128 * 256, p, 128, 256, 256, 256);            p += U1P;
        rp(u2w_a + (size_t)blk * 384 * 128, p, 384, 128, 128, 128);
    }
    rp(out1_w, o1p, 128, 128, 128, 128);
    rp(out2_w, o2p, 128, 128, 128, 128);

    init_s_kernel<<<(NN * HD + 255) / 256, 256, 0, stream>>>(z, emb, s, s_bf);
    init_v_kernel<<<(NN * HD + 255) / 256, 256, 0, stream>>>(vf, vproj_w, v);
    init_evec_kernel<<<(NE + 255) / 256, 256, 0, stream>>>(ei, pos, evec);

    for (int blk = 0; blk < NBLK; ++blk) {
        __bf16* p = blkp + (size_t)blk * TPB;
        __bf16 *s1p = p, *s2p = s1p + S1P, *e1p = s2p + S2P, *e2p = e1p + E1P;
        __bf16 *fsp = e2p + E2P, *fvp = fsp + FSP, *Upk = fvp + FVP, *Vpk = Upk + UP;
        __bf16 *u1p = Vpk + VP, *u2p = u1p + U1P;

        zero2_kernel<<<2048, 256, 0, stream>>>(dsA, (long)NN * HD, dvA,
                                               (long)NN * HD * 3);
        edge_msg_kernel<<<NE / 16, 128, 0, stream>>>(
            s_bf, v, evec, ei, ei + NE, eattr,
            s1p, s1b_a + (size_t)blk * HD,
            s2p, s2b_a + (size_t)blk * 3 * HD,
            e1p, e1b_a + (size_t)blk * HD,
            e2p, e2b_a + (size_t)blk * 3 * HD,
            dsA, dvA);
        node_update_kernel<<<NN / 16, 128, 0, stream>>>(
            s, s_bf, v, dsA, dvA,
            fsp, fsb_a + (size_t)blk * HD, fvp, Upk, Vpk,
            u1p, u1b_a + (size_t)blk * HD,
            u2p, u2b_a + (size_t)blk * 3 * HD);
    }

    out_mlp_kernel<<<NN / 16, 128, 0, stream>>>(s_bf, o1p, out1_b, o2p, out2_b,
                                                (float*)d_out);
    copy_v_kernel<<<((long)NN * HD * 3 + 255) / 256, 256, 0, stream>>>(
        v, (float*)d_out + (size_t)NN * HD);
}
